// QuantizedConv2d_66494683676723
// MI455X (gfx1250) — 8-bit matrix operand order repaired and checked by reading, compiled, not run on hardware
//
#include <hip/hip_runtime.h>
#include <stdint.h>

typedef int v8i __attribute__((ext_vector_type(8)));
typedef int v4i __attribute__((ext_vector_type(4)));
typedef int v2i __attribute__((ext_vector_type(2)));

#define H_IMG 56
#define W_IMG 56
#define HW    3136
#define CIN   128
#define COUT  256
#define NB    32
#define NPIX  (NB*HW)

__global__ __launch_bounds__(256) void pack_input_kernel(
    const int* __restrict__ x, uint8_t* __restrict__ xp)
{
    int tid = blockIdx.x * blockDim.x + threadIdx.x;
    int pix = tid >> 5;
    int g   = tid & 31;
    if (pix >= NPIX) return;
    int n  = pix / HW;
    int hw = pix - n * HW;
    const int* src = x + (size_t)(n * CIN + g * 4) * HW + hw;
    unsigned int b0 = (unsigned int)(uint8_t)(src[0]      + 128);
    unsigned int b1 = (unsigned int)(uint8_t)(src[HW]     + 128);
    unsigned int b2 = (unsigned int)(uint8_t)(src[2 * HW] + 128);
    unsigned int b3 = (unsigned int)(uint8_t)(src[3 * HW] + 128);
    *(unsigned int*)(xp + (size_t)pix * CIN + g * 4) =
        b0 | (b1 << 8) | (b2 << 16) | (b3 << 24);
}

__global__ __launch_bounds__(256) void pack_weight_kernel(
    const int* __restrict__ w, int8_t* __restrict__ wpf, int* __restrict__ wk)
{
    int tid = blockIdx.x * blockDim.x + threadIdx.x;
    if (tid >= COUT * 9) return;
    int co = tid / 9, tap = tid % 9;
    int tile = co >> 4;
    int col  = co & 15;
    int s = 0;
    for (int ci = 0; ci < CIN; ++ci) {
        int v = w[(size_t)(co * CIN + ci) * 9 + tap];
        s += v;
        int half  = ci >> 6;
        int r     = ci & 63;
        int hi    = (r >> 4) & 1;
        int chunk = r >> 5;
        int b     = r & 15;
        int lane  = col + hi * 16;
        size_t off = (size_t)((tile * 9 + tap) * 2 + half) * 1024
                   + lane * 32 + chunk * 16 + b;
        wpf[off] = (int8_t)v;
    }
    wk[co * 9 + tap] = s;
}

__global__ __launch_bounds__(256) void conv_wmma_kernel(
    const uint8_t* __restrict__ xp,
    const int8_t*  __restrict__ wpf,
    const int*     __restrict__ wk,
    const int*     __restrict__ bias,
    const float*   __restrict__ in_s,
    const float*   __restrict__ w_s,
    const float*   __restrict__ out_s,
    const int*     __restrict__ in_zp,
    const int*     __restrict__ out_zp,
    float*         __restrict__ out)
{
    const int lane   = threadIdx.x & 31;
    const int wave   = threadIdx.x >> 5;
    const int m      = wave & 1;
    const int nn     = wave >> 1;
    const int pixT   = blockIdx.x * 32 + m * 16;
    const int coBase = nn * 64;
    const int tileB  = nn * 4;

    const int laneLo = lane & 15;
    const int hi16   = lane >> 4;

    const int apix = pixT + laneLo;
    const int an   = apix / HW;
    const int ahw  = apix - an * HW;
    const int ah   = ahw / W_IMG;
    const int aw   = ahw - ah * W_IMG;

    const int k0a = hi16 * 16;

    const int8_t* wfl = wpf + lane * 32;

    v8i acc0 = {}, acc1 = {}, acc2 = {}, acc3 = {};

    #pragma unroll
    for (int tap = 0; tap < 9; ++tap) {
        const int dh = tap / 3 - 1;
        const int dw = tap % 3 - 1;
        const int hh = ah + dh, ww = aw + dw;
        const bool valid = (hh >= 0) && (hh < H_IMG) && (ww >= 0) && (ww < W_IMG);
        const uint8_t* arow = xp + (size_t)(apix + dh * W_IMG + dw) * CIN;

        #pragma unroll
        for (int half = 0; half < 2; ++half) {
            v2i q0 = {}, q1 = {}, q2 = {}, q3 = {};
            if (valid) {
                const v2i* p = (const v2i*)(arow + half * 64 + k0a);
                q0 = p[0];
                q1 = p[1];
                q2 = p[4];
                q3 = p[5];
            }
            v8i a;
            a[0] = q0[0]; a[1] = q0[1];
            a[2] = q1[0]; a[3] = q1[1];
            a[4] = q2[0]; a[5] = q2[1];
            a[6] = q3[0]; a[7] = q3[1];

            {
                const int8_t* bp = wfl + (size_t)(((tileB + 0) * 9 + tap) * 2 + half) * 1024;
                v4i b0 = *(const v4i*)(bp);
                v4i b1 = *(const v4i*)(bp + 16);
                v8i b;
                b[0] = b0[0]; b[1] = b0[1]; b[2] = b0[2]; b[3] = b0[3];
                b[4] = b1[0]; b[5] = b1[1]; b[6] = b1[2]; b[7] = b1[3];
                acc0 = __builtin_amdgcn_wmma_i32_16x16x64_iu8(false, a, true, b, acc0, false, false);
            }
            {
                const int8_t* bp = wfl + (size_t)(((tileB + 1) * 9 + tap) * 2 + half) * 1024;
                v4i b0 = *(const v4i*)(bp);
                v4i b1 = *(const v4i*)(bp + 16);
                v8i b;
                b[0] = b0[0]; b[1] = b0[1]; b[2] = b0[2]; b[3] = b0[3];
                b[4] = b1[0]; b[5] = b1[1]; b[6] = b1[2]; b[7] = b1[3];
                acc1 = __builtin_amdgcn_wmma_i32_16x16x64_iu8(false, a, true, b, acc1, false, false);
            }
            {
                const int8_t* bp = wfl + (size_t)(((tileB + 2) * 9 + tap) * 2 + half) * 1024;
                v4i b0 = *(const v4i*)(bp);
                v4i b1 = *(const v4i*)(bp + 16);
                v8i b;
                b[0] = b0[0]; b[1] = b0[1]; b[2] = b0[2]; b[3] = b0[3];
                b[4] = b1[0]; b[5] = b1[1]; b[6] = b1[2]; b[7] = b1[3];
                acc2 = __builtin_amdgcn_wmma_i32_16x16x64_iu8(false, a, true, b, acc2, false, false);
            }
            {
                const int8_t* bp = wfl + (size_t)(((tileB + 3) * 9 + tap) * 2 + half) * 1024;
                v4i b0 = *(const v4i*)(bp);
                v4i b1 = *(const v4i*)(bp + 16);
                v8i b;
                b[0] = b0[0]; b[1] = b0[1]; b[2] = b0[2]; b[3] = b0[3];
                b[4] = b1[0]; b[5] = b1[1]; b[6] = b1[2]; b[7] = b1[3];
                acc3 = __builtin_amdgcn_wmma_i32_16x16x64_iu8(false, a, true, b, acc3, false, false);
            }
        }
    }

    const float scale = in_s[0] * w_s[0] / out_s[0];
    const float ozp   = (float)out_zp[0];
    const int   C     = 128 + in_zp[0];

    int rn[8], rh[8], rw[8];
    unsigned rmask[8];
    #pragma unroll
    for (int v = 0; v < 8; ++v) {
        const int pix = pixT + v + hi16 * 8;
        const int n   = pix / HW;
        const int hw  = pix - n * HW;
        const int h   = hw / W_IMG;
        const int w   = hw - h * W_IMG;
        rn[v] = n; rh[v] = h; rw[v] = w;
        unsigned msk = 0;
        #pragma unroll
        for (int t = 0; t < 9; ++t) {
            const int dh = t / 3 - 1, dw = t % 3 - 1;
            const int hh = h + dh, ww = w + dw;
            if (hh >= 0 && hh < H_IMG && ww >= 0 && ww < W_IMG) msk |= (1u << t);
        }
        rmask[v] = msk;
    }

    #pragma unroll
    for (int t4 = 0; t4 < 4; ++t4) {
        const int co = coBase + t4 * 16 + laneLo;
        const int bv = bias[co];
        int wkv[9];
        #pragma unroll
        for (int t = 0; t < 9; ++t) wkv[t] = wk[co * 9 + t];

        const v8i acc = (t4 == 0) ? acc0 : (t4 == 1) ? acc1 : (t4 == 2) ? acc2 : acc3;

        #pragma unroll
        for (int v = 0; v < 8; ++v) {
            int s = 0;
            #pragma unroll
            for (int t = 0; t < 9; ++t)
                if (rmask[v] & (1u << t)) s += wkv[t];

            const int a32 = acc[v] - C * s + bv;
            float f = rintf((float)a32 * scale + ozp);
            f = fminf(fmaxf(f, -128.0f), 127.0f);
            out[(size_t)(rn[v] * COUT + co) * HW + rh[v] * W_IMG + rw[v]] = f;
        }
    }
}

extern "C" void kernel_launch(void* const* d_in, const int* in_sizes, int n_in,
                              void* d_out, int out_size, void* d_ws, size_t ws_size,
                              hipStream_t stream)
{
    const int*   x      = (const int*)d_in[0];
    const int*   w      = (const int*)d_in[1];
    const int*   bias   = (const int*)d_in[2];
    const float* in_s   = (const float*)d_in[3];
    const float* w_s    = (const float*)d_in[4];
    const float* out_s  = (const float*)d_in[5];
    const int*   in_zp  = (const int*)d_in[6];
    const int*   out_zp = (const int*)d_in[7];
    float*       out    = (float*)d_out;

    uint8_t* xp  = (uint8_t*)d_ws;
    int8_t*  wpf = (int8_t*)((char*)d_ws + 12845056);
    int*     wk  = (int*)((char*)d_ws + 12845056 + 294912);

    pack_input_kernel<<<(NPIX * 32) / 256, 256, 0, stream>>>(x, xp);
    pack_weight_kernel<<<(COUT * 9 + 255) / 256, 256, 0, stream>>>(w, wpf, wk);

    conv_wmma_kernel<<<dim3(NPIX / 32), 256, 0, stream>>>(
        xp, wpf, wk, bias, in_s, w_s, out_s, in_zp, out_zp, out);
}
